// MultiboxLoss_28853590294774
// MI455X (gfx1250) — compile-verified, matrix operand order checked by reading, not run on hardware
//
#include <hip/hip_runtime.h>
#include <hip/hip_bf16.h>

// ---------------------------------------------------------------------------
// MultiboxLoss on MI455X (gfx1250, wave32).
// Dominant cost: 85MB read of pred_confs (log_softmax) -> memory bound
// (~4us floor at 23.3 TB/s). Conf tiles are staged cache->LDS with the CDNA5
// async-to-LDS path (no VGPR round trip, ASYNCcnt-tracked), and the exp
// row-sum reduction runs on the matrix pipe (V_WMMA_F32_16X16X32_F16 with a
// ones B-matrix) so VALU/TRANS stay free for exp/log work.
// ---------------------------------------------------------------------------

typedef __attribute__((ext_vector_type(16))) _Float16 v16h;
typedef __attribute__((ext_vector_type(8)))  float    v8f;

#define NP 262144
#define NG 256
#define NC 81
#define THRESH 0.5f

// workspace layout (bytes)
#define OFF_SCAL 0u                          // 64B: huber_sum,num_pos,ce_pos,nsel,ce_neg
#define OFF_INTS 64u                         // NP floats
#define OFF_NLL  (OFF_INTS + NP*4u)          // NP floats
#define OFF_MASK (OFF_NLL  + NP*4u)          // NP bytes
#define OFF_HIST (OFF_MASK + NP)             // 65536 u32
#define OFF_BCNT (OFF_HIST + 65536u*4u)      // 1024 u32
#define OFF_BOFF (OFF_BCNT + 1024u*4u)       // 1024 u32

__device__ __forceinline__ float huber1(float x) {
  float a = fabsf(x);
  return (a < 1.0f) ? 0.5f * x * x : a - 0.5f;
}

// --- Kernel A: pairwise overlap vs 256 GT (GT in LDS), per-prior reductions -
__global__ void mb_overlap(const float4* __restrict__ gt,
                           const float4* __restrict__ pred,
                           float* __restrict__ inter_sums,
                           unsigned char* __restrict__ posmask,
                           float* __restrict__ huber_sum,
                           unsigned* __restrict__ num_pos) {
  __shared__ float4 g[NG];
  __shared__ float  hred[256];
  __shared__ unsigned cred[256];
  int tid = threadIdx.x;
  g[tid] = gt[tid];                 // 256 threads == 256 GT boxes (4KB)
  __syncthreads();

  int i = blockIdx.x * 256 + tid;
  float4 p = pred[i];
  float L = p.x, T = p.y, W = p.z, H = p.w;
  float B = L + W, R = T + H;       // reference's (swapped) convention

  float sum = 0.f, bestv = -1.f;
  int bestj = 0;
  bool pos = false;
#pragma unroll 4
  for (int j = 0; j < NG; ++j) {
    float4 q = g[j];
    float gr = q.y + q.w, gb = q.x + q.z;
    float xov = fminf(R, gr) - fmaxf(L, q.x); xov = fmaxf(xov, 0.f);
    float yov = fminf(B, gb) - fmaxf(T, q.y); yov = fmaxf(yov, 0.f);
    float inter = xov * yov;
    bool  m  = inter > THRESH;
    float mv = m ? inter : 0.f;
    sum += mv;
    if (mv > bestv) { bestv = mv; bestj = j; }
    pos |= m;
  }
  inter_sums[i] = sum;
  posmask[i] = pos ? 1u : 0u;

  float hub = 0.f;
  if (pos) {
    float4 q = g[bestj];
    float g0 = (q.x - L) / W;
    float g1 = (q.y - T) / H;
    float g2 = __logf(q.z / W);
    float g3 = __logf(q.w / H);
    hub = huber1(g0 - q.x) + huber1(g1 - q.y) + huber1(g2 - q.z) + huber1(g3 - q.w);
  }
  hred[tid] = hub; cred[tid] = pos ? 1u : 0u;
  __syncthreads();
  for (int s = 128; s > 0; s >>= 1) {
    if (tid < s) { hred[tid] += hred[tid + s]; cred[tid] += cred[tid + s]; }
    __syncthreads();
  }
  if (tid == 0) { atomicAdd(huber_sum, hred[0]); atomicAdd(num_pos, cred[0]); }
}

// --- Kernel B: NLL via logsumexp; async cache->LDS staging + WMMA row-sums --
// One wave handles 16 priors. A = 16x32 f16 tile of exp(x-max) (zero-padded
// beyond 81 classes), B = ones => D rows hold the per-prior exp-sums.
__global__ void mb_nll(const float* __restrict__ conf,
                       const int* __restrict__ gt_ids,
                       const unsigned char* __restrict__ posmask,
                       float* __restrict__ nll,
                       float* __restrict__ ce_pos_sum) {
  __shared__ float lconf[128 * NC];           // 41,472 B (<< 320KB WGP LDS)
  int tid = threadIdx.x;
  int tile = blockIdx.x * 128;
  const float* src = conf + (size_t)tile * NC;

  // Aliasing anchor: escape lconf's address so the async asm's "memory"
  // clobber is known to (possibly) write the lconf object. Reads below go
  // through `lconf` directly so they stay addrspace(3) -> ds_load_*.
  float* lbase = lconf;
  asm volatile("" : "+v"(lbase) :: "memory");

  // Stage the 128x81 f32 tile with GLOBAL_LOAD_ASYNC_TO_LDS_B128: direct
  // cache->LDS, tracked by ASYNCcnt (no VGPR round trip on the hot 85MB
  // stream). VDST operand = LDS byte offset = low 32 bits of the generic
  // pointer (flat-aperture mapping).
  for (int idx4 = tid; idx4 < (128 * NC) / 4; idx4 += 256) {
    unsigned lds_off = (unsigned)(size_t)(lbase + idx4 * 4);
    const float* gaddr = src + idx4 * 4;
    asm volatile("global_load_async_to_lds_b128 %0, %1, off"
                 :: "v"(lds_off), "v"(gaddr) : "memory");
  }
  if (blockIdx.x + 1 < gridDim.x)
    __builtin_prefetch(src + 128 * NC + tid * 32, 0, 1);   // global_prefetch_b8
  asm volatile("s_wait_asynccnt 0x0" ::: "memory");        // my async ops done
  __syncthreads();                                         // all waves' visible

  int wave = tid >> 5, lane = tid & 31;
  int r = lane & 15, h = lane >> 4;           // row within tile-of-16, K-half
  const int rowbase = (wave * 16 + r) * NC;   // index into lconf (ds_load path)

  // pass 1: row max (two lanes per row, combine across the half-wave split)
  float m = -INFINITY;
  for (int k = h; k < NC; k += 2) m = fmaxf(m, lconf[rowbase + k]);
  m = fmaxf(m, __shfl_xor(m, 16, 32));

  // pass 2: sum(exp(x-m)) via 3x V_WMMA_F32_16X16X32_F16 with ones-B
  v16h b;
#pragma unroll
  for (int e = 0; e < 16; ++e) b[e] = (_Float16)1.0f;
  v8f c = {};
  for (int chunk = 0; chunk < 3; ++chunk) {
    v16h a;
#pragma unroll
    for (int e = 0; e < 16; ++e) {
      // 16-bit A layout: elems 0..7 -> K = 8h+e ; elems 8..15 -> K = 16+8h+(e-8)
      int K = (e < 8) ? (8 * h + e) : (16 + 8 * h + (e - 8));
      int k = chunk * 32 + K;
      float x = (k < NC) ? __expf(lconf[rowbase + k] - m) : 0.f;
      a[e] = (_Float16)x;
    }
    c = __builtin_amdgcn_wmma_f32_16x16x32_f16(false, a, false, b,
                                               (short)0, c, false, false);
  }
  // D layout: lane holds rows {8h..8h+7} in c[0..7]; fetch own row via shfl
  float carr[8];
#pragma unroll
  for (int v = 0; v < 8; ++v) carr[v] = c[v];
  float val = carr[r & 7];                     // rowsum of (r&7) + 8h
  int srcLane = (r < 8) ? r : (r + 8);
  float s = __shfl(val, srcLane, 32);

  if (h == 0) {
    int i = tile + wave * 16 + r;
    int gid = gt_ids[i];
    float v = m + __logf(s) - lconf[rowbase + gid];   // -log_softmax[gid]
    nll[i] = v;
    if (posmask[i]) atomicAdd(ce_pos_sum, v);
  }
}

// --- Hard-negative machinery: rank(v) via 64K-bin histogram of float bits ---
__global__ void mb_hist(const float* __restrict__ inter_sums,
                        unsigned* __restrict__ hist) {
  int i = blockIdx.x * 256 + threadIdx.x;
  unsigned bits = __float_as_uint(inter_sums[i]);   // values >= 0: bit order = value order
  atomicAdd(&hist[bits >> 16], 1u);
}

__global__ void mb_scan(unsigned* __restrict__ hist) {  // 1 block x 256
  __shared__ unsigned csum[256];
  int t = threadIdx.x;
  unsigned s = 0;
  for (int b = 0; b < 256; ++b) s += hist[t * 256 + b];
  csum[t] = s;
  __syncthreads();
  if (t == 0) {
    unsigned run = 0;
    for (int k = 0; k < 256; ++k) { unsigned v = csum[k]; csum[k] = run; run += v; }
  }
  __syncthreads();
  unsigned run = csum[t];
  for (int b = 0; b < 256; ++b) { unsigned v = hist[t * 256 + b]; hist[t * 256 + b] = run; run += v; }
}

__device__ __forceinline__ bool neg_flag(const float* inter_sums,
                                         const unsigned* hist, int i) {
  unsigned bits = __float_as_uint(inter_sums[i]);
  return hist[bits >> 16] <= (unsigned)i;    // cntLess(v) <= i  <=>  v <= sorted[i]
}

__global__ void mb_negcount(const float* __restrict__ inter_sums,
                            const unsigned* __restrict__ hist,
                            unsigned* __restrict__ bcnt) {
  __shared__ unsigned red[8];
  int tid = threadIdx.x;
  int i = blockIdx.x * 256 + tid;
  bool neg = neg_flag(inter_sums, hist, i);
  unsigned long long ball = __ballot(neg);   // wave32: low 32 bits valid
  int lane = tid & 31, wave = tid >> 5;
  if (lane == 0) red[wave] = (unsigned)__popcll(ball);
  __syncthreads();
  if (tid == 0) {
    unsigned s = 0;
    for (int w = 0; w < 8; ++w) s += red[w];
    bcnt[blockIdx.x] = s;
  }
}

__global__ void mb_bscan(const unsigned* __restrict__ bcnt,
                         unsigned* __restrict__ boff) {   // 1 thread
  if (threadIdx.x == 0 && blockIdx.x == 0) {
    unsigned run = 0;
    for (int b = 0; b < NP / 256; ++b) { boff[b] = run; run += bcnt[b]; }
  }
}

__global__ void mb_select(const float* __restrict__ inter_sums,
                          const unsigned* __restrict__ hist,
                          const unsigned* __restrict__ boff,
                          const float* __restrict__ nll,
                          const unsigned* __restrict__ num_pos,
                          float* __restrict__ ce_neg_sum,
                          unsigned* __restrict__ nsel) {
  __shared__ unsigned wsum[8];
  int tid = threadIdx.x;
  int i = blockIdx.x * 256 + tid;
  bool neg = neg_flag(inter_sums, hist, i);
  unsigned long long ball = __ballot(neg);
  int lane = tid & 31, wave = tid >> 5;
  if (lane == 0) wsum[wave] = (unsigned)__popcll(ball);
  __syncthreads();
  unsigned woff = 0;
  for (int w = 0; w < wave; ++w) woff += wsum[w];
  unsigned incl = woff + (unsigned)__popcll(ball & ((2ull << lane) - 1ull));
  unsigned rank = boff[blockIdx.x] + incl;   // inclusive cumsum when neg
  unsigned hard = 3u * (*num_pos);
  if (neg && rank <= hard) {
    atomicAdd(ce_neg_sum, nll[i]);
    atomicAdd(nsel, 1u);
  }
}

__global__ void mb_final(const float* __restrict__ scal_f,
                         const unsigned* __restrict__ scal_u,
                         float* __restrict__ out) {
  if (threadIdx.x == 0 && blockIdx.x == 0) {
    float hub = scal_f[0];                 // huber sum
    float np  = (float)scal_u[1];          // num_pos
    float cep = scal_f[2];                 // ce_pos numerator
    unsigned nselv = scal_u[3];
    float cen = scal_f[4];                 // ce_neg numerator
    float ns = (float)(nselv > 1u ? nselv : 1u);
    out[0] = cep / np + cen / ns + hub / (4.f * np * np);
  }
}

extern "C" void kernel_launch(void* const* d_in, const int* in_sizes, int n_in,
                              void* d_out, int out_size, void* d_ws, size_t ws_size,
                              hipStream_t stream) {
  const float4* gt_locs   = (const float4*)d_in[0];
  const float4* pred_locs = (const float4*)d_in[1];
  const int*    gt_ids    = (const int*)d_in[2];
  const float*  pred_confs= (const float*)d_in[3];
  float* out = (float*)d_out;

  char* w = (char*)d_ws;
  float*    scal_f = (float*)(w + OFF_SCAL);
  unsigned* scal_u = (unsigned*)(w + OFF_SCAL);
  float*    inter_sums = (float*)(w + OFF_INTS);
  float*    nll        = (float*)(w + OFF_NLL);
  unsigned char* mask  = (unsigned char*)(w + OFF_MASK);
  unsigned* hist = (unsigned*)(w + OFF_HIST);
  unsigned* bcnt = (unsigned*)(w + OFF_BCNT);
  unsigned* boff = (unsigned*)(w + OFF_BOFF);

  // deterministic zero-init of accumulators + histogram (capture-safe)
  hipMemsetAsync(w + OFF_SCAL, 0, 64, stream);
  hipMemsetAsync(w + OFF_HIST, 0, 65536u * 4u, stream);

  mb_overlap<<<NP / 256, 256, 0, stream>>>(gt_locs, pred_locs, inter_sums, mask,
                                           &scal_f[0], &scal_u[1]);
  mb_nll<<<NP / 128, 256, 0, stream>>>(pred_confs, gt_ids, mask, nll, &scal_f[2]);
  mb_hist<<<NP / 256, 256, 0, stream>>>(inter_sums, hist);
  mb_scan<<<1, 256, 0, stream>>>(hist);
  mb_negcount<<<NP / 256, 256, 0, stream>>>(inter_sums, hist, bcnt);
  mb_bscan<<<1, 32, 0, stream>>>(bcnt, boff);
  mb_select<<<NP / 256, 256, 0, stream>>>(inter_sums, hist, boff, nll,
                                          &scal_u[1], &scal_f[4], &scal_u[3]);
  mb_final<<<1, 32, 0, stream>>>(scal_f, scal_u, out);
}
